// BipartiteGraphAttentionLayer_24807731102053
// MI455X (gfx1250) — matrix operand order repaired and checked by reading, not compiled, not run on hardware
//
#include <hip/hip_runtime.h>

// ---------------------------------------------------------------------------
// Bipartite GAT layer for MI455X (gfx1250, wave32, WMMA + async-LDS DMA).
//
// Roofline: ~22 GFLOP vs ~30MB HBM traffic -> compute bound; all GEMM work
// routed through v_wmma_f32_16x16x32_f16 (f16 in, f32 accumulate).
// Attention logits are rank-structured (logit = a_src[n]+a_dst[m]), so the
// N x N matrix is generated in registers (flash-softmax) and only P @ h_dst
// hits the matrix pipes. h_dst^T is staged into LDS with
// GLOBAL_LOAD_ASYNC_TO_LDS_B128 (ASYNCcnt), double-buffered so the 128KB DMA
// of stage s+1 overlaps the WMMA/VALU compute of stage s; ~267KB of the
// 320KB WGP LDS used. Rows carry a +8-half skew -> conflict-free b128 reads.
// ---------------------------------------------------------------------------

#define NBATCH 4
#define NHEAD  4
#define INDIM  256
#define ODIM   64
#define NSRC   2048
#define NDST   2048
#define LEAKY  0.2f

#define DSTG    1024          // dst nodes staged in LDS per pass
#define HSTRIDE (DSTG + 8)    // +8 halfs row skew -> 4-bank rotation
#define NSTAGE  (NDST / DSTG) // 2
#define BUFH    ((size_t)ODIM * HSTRIDE)   // halfs per LDS buffer

typedef __attribute__((ext_vector_type(16))) _Float16 v16h;
typedef __attribute__((ext_vector_type(8)))  float    v8f;

union AF { v16h v; unsigned int u[8]; _Float16 e[16]; };
union BF { v16h v; uint4 q[2]; };

__device__ __forceinline__ v8f wmma_f16(v16h a, v16h b, v8f c) {
  // D = A(16x32 f16) * B(32x16 f16) + C(16x16 f32)
  return __builtin_amdgcn_wmma_f32_16x16x32_f16(false, a, false, b,
                                                (short)0, c, false, false);
}

// ---------------------------------------------------------------------------
// Stage A: projections.
//   h = feat @ W      (per head)  -> h_dst stored transposed f16 [b][h][o][node]
//   a = tanh(h) . w_att            -> a_src / a_dst  (deterministic reduction)
//   gate = sigmoid(feat_src @ H_w^T + H_b)   (src only)
// One workgroup = 16 node rows of one batch; 8 waves cover the 16 projection
// tiles (+16 gate tiles for src); K loop = 256/32 WMMAs per tile.
// ---------------------------------------------------------------------------
__global__ __launch_bounds__(256) void proj_kernel(
    const float* __restrict__ feat, const float* __restrict__ Wp,
    const float* __restrict__ watt, const float* __restrict__ Hw,
    const float* __restrict__ Hb, _Float16* __restrict__ hT,
    float* __restrict__ aout, float* __restrict__ gate,
    int is_src, int Nn)
{
  __shared__ _Float16 sfeat[16 * 264];   // 16 rows x 256 K, stride 264 (skewed)
  __shared__ float    apart[16][16];     // per-col-tile partial a, [tile][row]

  const int tid = threadIdx.x;
  const int nrt = Nn / 16;
  const int b   = blockIdx.x / nrt;
  const int rt  = blockIdx.x % nrt;
  const int row_base = rt * 16;

  // stage 16x256 feat tile as f16
  for (int i = tid; i < 16 * 256; i += 256) {
    const int r = i >> 8, k = i & 255;
    sfeat[r * 264 + k] =
        (_Float16)feat[((size_t)b * Nn + row_base + r) * INDIM + k];
  }
  __syncthreads();

  const int lane = tid & 31, wid = tid >> 5;
  const int n = lane & 15, hs = lane >> 4;

  // ---- projection tiles (ct = 0..15 -> col = head*64 + o) ----
  for (int ct = wid; ct < 16; ct += 8) {
    const int h = ct >> 2;
    const int o = ((ct & 3) << 4) + n;
    v8f acc = {};
    for (int kc = 0; kc < 8; ++kc) {
      const int k0 = kc * 32;
      // A fragment: elem i -> K = i + 16*hs (matches B)
      AF a;
#pragma unroll
      for (int j = 0; j < 8; ++j) {
        const int kp = 2 * j + hs * 16;
        a.u[j] = *(const unsigned int*)&sfeat[n * 264 + k0 + kp];
      }
      // B fragment (contiguous K order, same as A): lane col = n, elem i -> K = i + 16*hs
      AF bb;
      const float* src = &Wp[((size_t)h * INDIM + k0 + 16 * hs) * ODIM + o];
#pragma unroll
      for (int i = 0; i < 16; ++i) bb.e[i] = (_Float16)src[(size_t)i * ODIM];
      acc = wmma_f16(a.v, bb.v, acc);
    }
    // acc[v] = C[row = v + 8*hs][col = n]
    if (!is_src) {
      // store h_dst transposed f16: [b][h][o][node], packed pair stores
      const size_t base =
          (((size_t)b * NHEAD + h) * ODIM + o) * (size_t)NDST + row_base + 8 * hs;
#pragma unroll
      for (int m = 0; m < 4; ++m) {
        union { unsigned int u; _Float16 e[2]; } pk;
        pk.e[0] = (_Float16)acc[2 * m];
        pk.e[1] = (_Float16)acc[2 * m + 1];
        *(unsigned int*)&hT[base + 2 * m] = pk.u;
      }
    }
    // a partial: sum_o tanh(h[row,o]) * w[o]  (reduce 16 cols via shuffles)
    const float wv = watt[h * ODIM + o];
#pragma unroll
    for (int v = 0; v < 8; ++v) {
      float p = tanhf(acc[v]) * wv;
      p += __shfl_xor(p, 1);
      p += __shfl_xor(p, 2);
      p += __shfl_xor(p, 4);
      p += __shfl_xor(p, 8);
      if (n == 0) apart[ct][8 * hs + v] = p;
    }
  }

  // ---- gate tiles (src only, ct = 16..31 -> gate col j) ----
  if (is_src) {
    for (int ct = wid + 16; ct < 32; ct += 8) {
      const int j = ((ct - 16) << 4) + n;
      v8f acc = {};
      for (int kc = 0; kc < 8; ++kc) {
        const int k0 = kc * 32;
        AF a;
#pragma unroll
        for (int jj = 0; jj < 8; ++jj) {
          const int kp = 2 * jj + hs * 16;
          a.u[jj] = *(const unsigned int*)&sfeat[n * 264 + k0 + kp];
        }
        AF bb;  // B[k][j] = H_w[j][k] -> contiguous float4 loads
        const float4* src = (const float4*)&Hw[(size_t)j * INDIM + k0 + 16 * hs];
#pragma unroll
        for (int q = 0; q < 4; ++q) {
          const float4 f = src[q];
          bb.e[4 * q + 0] = (_Float16)f.x;
          bb.e[4 * q + 1] = (_Float16)f.y;
          bb.e[4 * q + 2] = (_Float16)f.z;
          bb.e[4 * q + 3] = (_Float16)f.w;
        }
        acc = wmma_f16(a.v, bb.v, acc);
      }
      const float hb = Hb[j];
#pragma unroll
      for (int v = 0; v < 8; ++v) {
        const int node = row_base + 8 * hs + v;
        const float g = 1.0f / (1.0f + __expf(-(acc[v] + hb)));
        gate[((size_t)b * Nn + node) * INDIM + j] = g;
      }
    }
  }

  __syncthreads();
  // deterministic cross-tile reduction: a[row] = sum over the head's 4 tiles
  if (tid < NHEAD * 16) {
    const int h = tid >> 4, row = tid & 15;
    float s = apart[h * 4 + 0][row] + apart[h * 4 + 1][row] +
              apart[h * 4 + 2][row] + apart[h * 4 + 3][row];
    aout[((size_t)b * NHEAD + h) * Nn + row_base + row] = s;
  }
}

// ---------------------------------------------------------------------------
// Stage B: flash-softmax attention + fused ELU/gate/residual epilogue.
// Workgroup = one (b,h) and 128 src rows (8 waves x 16 rows). h_dst^T is
// DMA'd into a double-buffered LDS tile with global_load_async_to_lds_b128;
// stage s+1's DMA overlaps stage s's compute, fenced by s_wait_asynccnt.
// ---------------------------------------------------------------------------
__global__ __launch_bounds__(256) void attn_kernel(
    const float* __restrict__ a_src, const float* __restrict__ a_dst,
    const _Float16* __restrict__ hT, const float* __restrict__ gate,
    const float* __restrict__ feat_src, const float* __restrict__ bias,
    float* __restrict__ out)
{
  extern __shared__ char smem[];
  _Float16* sh  = (_Float16*)smem;                       // 2 x 64 x HSTRIDE
  float*    sad = (float*)(smem + 2 * BUFH * 2);         // a_dst[2048]
  float*    sbc = sad + NDST;                            // 8x16 bcast

  const int tid  = threadIdx.x;
  const int bh   = blockIdx.x >> 4;      // b*NHEAD + h
  const int rblk = blockIdx.x & 15;
  const int b    = bh >> 2, h = bh & 3;
  const int lane = tid & 31, wid = tid >> 5;
  const int n = lane & 15, hs = lane >> 4;
  const int row = rblk * 128 + wid * 16 + n;   // this lane's P row (src node)

  const uint4* gsrc = (const uint4*)&hT[(size_t)bh * ODIM * NDST];

  // issue async DMA (memory -> LDS, ASYNCcnt) for one h_dst^T stage
  auto stage_async = [&](int s, _Float16* dstbuf) {
    for (int i = tid; i < ODIM * (DSTG / 8); i += 256) {
      const int o = i >> 7, p = i & 127;           // DSTG/8 = 128 uint4 per row
      const uint4* gp = &gsrc[(size_t)o * (NDST / 8) + s * (DSTG / 8) + p];
      const unsigned la =
          (unsigned)(size_t)&dstbuf[(size_t)o * HSTRIDE + p * 8];
      asm volatile("global_load_async_to_lds_b128 %0, %1, off"
                   :: "v"(la), "v"(gp) : "memory");
    }
  };

  // stage a_dst for this (b,h)
  for (int i = tid; i < NDST / 4; i += 256)
    ((float4*)sad)[i] = ((const float4*)&a_dst[(size_t)bh * NDST])[i];

  const float as = a_src[(size_t)bh * NSRC + row];

  float m_run = -3.0e38f, l_run = 0.0f;
  v8f acc[4];
#pragma unroll
  for (int t = 0; t < 4; ++t) acc[t] = {};

  stage_async(0, sh);
  asm volatile("s_wait_asynccnt 0x0" ::: "memory");
  __syncthreads();

  for (int s = 0; s < NSTAGE; ++s) {
    _Float16* cur = sh + (size_t)(s & 1) * BUFH;
    if (s + 1 < NSTAGE)                       // overlap next DMA with compute
      stage_async(s + 1, sh + (size_t)((s + 1) & 1) * BUFH);

    for (int c = 0; c < DSTG / 32; ++c) {
      const int k0 = s * DSTG + c * 32;   // global dst base of chunk
      const int kl = c * 32;              // LDS-local dst base

      // logits (leaky-relu) in the WMMA A layout; chunk max
      float lr[16], cm = -3.0e38f;
#pragma unroll
      for (int i = 0; i < 16; ++i) {
        const int K = i + hs * 16;
        float t = as + sad[k0 + K];
        t = t > 0.0f ? t : LEAKY * t;
        lr[i] = t;
        cm = fmaxf(cm, t);
      }
      cm = fmaxf(cm, __shfl_xor(cm, 16));      // join both K-halves of the row
      const float newm  = fmaxf(m_run, cm);
      const float scale = __expf(m_run - newm);

      AF a;
      float rs = 0.0f;
#pragma unroll
      for (int i = 0; i < 16; ++i) {
        const float p = __expf(lr[i] - newm);
        rs += p;
        a.e[i] = (_Float16)p;
      }
      rs += __shfl_xor(rs, 16);
      l_run = l_run * scale + rs;
      m_run = newm;

      // broadcast per-row rescale factor to accumulator-row mapping (v + 8*hs)
      if (hs == 0) sbc[wid * 16 + n] = scale;   // wave-private region, in-order DS
      __builtin_amdgcn_wave_barrier();
      float sc[8];
#pragma unroll
      for (int v = 0; v < 8; ++v) sc[v] = sbc[wid * 16 + v + 8 * hs];
#pragma unroll
      for (int t = 0; t < 4; ++t)
#pragma unroll
        for (int v = 0; v < 8; ++v) acc[t][v] *= sc[v];

      // 4 column tiles of h_dst: contiguous b128 LDS loads (B layout K = i + 16*hs)
#pragma unroll
      for (int t = 0; t < 4; ++t) {
        BF bf;
        const uint4* ps =
            (const uint4*)&cur[(size_t)(t * 16 + n) * HSTRIDE + kl + 16 * hs];
        bf.q[0] = ps[0];
        bf.q[1] = ps[1];
        acc[t] = wmma_f16(a.v, bf.v, acc[t]);
      }
    }

    if (s + 1 < NSTAGE) {                 // fence own DMA, then WG-wide barrier
      asm volatile("s_wait_asynccnt 0x0" ::: "memory");
      __syncthreads();
    }
  }

  // broadcast per-row softmax denominator
  if (hs == 0) sbc[wid * 16 + n] = l_run;
  __builtin_amdgcn_wave_barrier();
  float linv[8];
#pragma unroll
  for (int v = 0; v < 8; ++v) linv[v] = 1.0f / sbc[wid * 16 + v + 8 * hs];

  // epilogue: +bias, ELU, gated residual with feat_src
#pragma unroll
  for (int t = 0; t < 4; ++t) {
    const float bo = bias[t * 16 + n];
#pragma unroll
    for (int v = 0; v < 8; ++v) {
      const int node = rblk * 128 + wid * 16 + 8 * hs + v;
      const int col  = h * ODIM + t * 16 + n;
      float fo = acc[t][v] * linv[v] + bo;
      fo = fo > 0.0f ? fo : (__expf(fo) - 1.0f);
      const size_t gi = ((size_t)b * NSRC + node) * INDIM + col;
      const float g  = gate[gi];
      const float fs = feat_src[gi];
      out[gi] = g * fo + (1.0f - g) * fs;
    }
  }
}

// ---------------------------------------------------------------------------
extern "C" void kernel_launch(void* const* d_in, const int* in_sizes, int n_in,
                              void* d_out, int out_size, void* d_ws, size_t ws_size,
                              hipStream_t stream) {
  (void)in_sizes; (void)n_in; (void)out_size; (void)ws_size;
  const float* feat_src = (const float*)d_in[0];  // [B,2048,256]
  const float* feat_dst = (const float*)d_in[1];  // [B,2048,256]
  const float* Wp       = (const float*)d_in[2];  // [4,256,64]
  const float* bvec     = (const float*)d_in[3];  // [64]
  const float* w_src    = (const float*)d_in[4];  // [4,64]
  const float* w_dst    = (const float*)d_in[5];  // [4,64]
  const float* Hw       = (const float*)d_in[6];  // [256,256]
  const float* Hb       = (const float*)d_in[7];  // [256]
  float* out = (float*)d_out;

  // workspace layout (~12.3 MB)
  char* ws = (char*)d_ws;
  _Float16* hdstT = (_Float16*)ws;                       // [B][H][64][2048] f16
  size_t off = (size_t)NBATCH * NHEAD * ODIM * NDST * sizeof(_Float16);
  float* a_src = (float*)(ws + off); off += (size_t)NBATCH * NHEAD * NSRC * 4;
  float* a_dst = (float*)(ws + off); off += (size_t)NBATCH * NHEAD * NDST * 4;
  float* gate  = (float*)(ws + off);

  // Stage A: dst projection (h_dst^T f16 + a_dst), then src (a_src + gate)
  proj_kernel<<<dim3(NBATCH * (NDST / 16)), dim3(256), 0, stream>>>(
      feat_dst, Wp, w_dst, nullptr, nullptr, hdstT, a_dst, nullptr, 0, NDST);
  proj_kernel<<<dim3(NBATCH * (NSRC / 16)), dim3(256), 0, stream>>>(
      feat_src, Wp, w_src, Hw, Hb, nullptr, a_src, gate, 1, NSRC);

  // Stage B: flash attention + epilogue.
  // Dynamic LDS: 2 x 132,096 (double-buffered h_dst^T) + 8,192 + 512 = 272,896 B
  const size_t smem_b = 2 * BUFH * 2 + (size_t)NDST * 4 + 8 * 16 * 4;
  attn_kernel<<<dim3(NBATCH * NHEAD * (NSRC / 128)), dim3(256), smem_b, stream>>>(
      a_src, a_dst, hdstT, gate, feat_src, bvec, out);
}
